// HierarchicalAttention_60730837566396
// MI455X (gfx1250) — matrix operand order repaired and checked by reading, not compiled, not run on hardware
//
#include <hip/hip_runtime.h>

typedef __attribute__((ext_vector_type(16))) __bf16 v16bf;
typedef __attribute__((ext_vector_type(8)))  float  v8f;
typedef __attribute__((ext_vector_type(4)))  unsigned int u32x4;

struct U32x8 { u32x4 lo; u32x4 hi; };

static __device__ __forceinline__ v16bf make_frag(u32x4 lo, u32x4 hi) {
  U32x8 u{lo, hi};
  return __builtin_bit_cast(v16bf, u);
}

// f32 -> bf16 round-to-nearest-even (bit pattern)
static __device__ __forceinline__ unsigned short f2bf(float f) {
  unsigned int u = __builtin_bit_cast(unsigned int, f);
  unsigned int r = u + 0x7FFFu + ((u >> 16) & 1u);
  return (unsigned short)(r >> 16);
}

static __device__ __forceinline__ float fast_tanh(float x) {
  x = fminf(15.f, fmaxf(-15.f, x));
  float e = __expf(2.f * x);
  return (e - 1.f) / (e + 1.f);
}

// ---------------- problem constants ----------------
// B=8, P=120, N_SEN=16, L=128, H=768, E=960, M=E*L=122880
#define HH 768
#define LL 128
#define EE 960

// d_out element offsets (f32), in reference return order
#define OUT_MIX       0                        // 960*2*768   = 1474560
#define OUT_FINAL     1474560                  // 8*16*768    = 98304
#define OUT_CLSMAT    1572864                  // 8*16*16*768 = 1572864
#define OUT_CLSSCORE  3145728                  // 960*2       = 1920
#define OUT_SCMAT     3147648                  // 4096
#define OUT_H1MAT     3151744                  // 4096
#define OUT_H2MAT     3155840                  // 4096
#define OUT_TOTAL     3159936
#define ZERO_N        (OUT_TOTAL - OUT_CLSMAT) // 1587072

// ---------------- kernel 1: W_st (f32, [k][n]) -> Wt bf16 transposed [n][k]
__global__ __launch_bounds__(256) void k_prep_wt(const float* __restrict__ Wst,
                                                 unsigned short* __restrict__ Wt) {
  int i = blockIdx.x * 256 + threadIdx.x;      // i = n*768 + k, i < 589824
  int n = i / HH, k = i % HH;
  Wt[i] = f2bf(Wst[k * HH + n]);
}

// ---------------- kernel 2: fused bf16-WMMA GEMM + tanh + dot(w_st2) -> tok_score
// M-partitioned: one block = 128 rows, wave wv owns rows [wv*16, wv*16+16) and
// sweeps ALL 48 N-tiles so the 8 waves share B fragments through WGP$.
// grid: 122880/128 = 960 blocks, 256 threads (8 waves)
__global__ __launch_bounds__(256) void k_score(const float* __restrict__ seq,
                                               const unsigned short* __restrict__ Wt,
                                               const float* __restrict__ b_st,
                                               const float* __restrict__ w_st2,
                                               const float* __restrict__ b_st2,
                                               float* __restrict__ tokws) {
  __shared__ unsigned int sA[128 * 388];  // 128 rows x 768 bf16, padded to 388 dwords/row (~194KB)
  const int t = threadIdx.x;
  const int m0 = blockIdx.x * 128;

  // stage 128x768 f32 -> bf16 LDS
  const float4* sp = (const float4*)(seq + (size_t)m0 * HH);
#pragma unroll
  for (int i = 0; i < 96; ++i) {
    int c = t + i * 256;                  // float4 chunk index, 0..24575
    int row = c / 192;
    int col4 = c % 192;
    float4 v = sp[c];
    unsigned int lo = ((unsigned int)f2bf(v.y) << 16) | f2bf(v.x);
    unsigned int hi = ((unsigned int)f2bf(v.w) << 16) | f2bf(v.z);
    int di = row * 388 + col4 * 2;
    sA[di] = lo; sA[di + 1] = hi;
  }
  __syncthreads();

  const int wv = t >> 5, lane = t & 31;
  const int col = lane & 15, half = lane >> 4;
  const float bst2v = b_st2[0];

  float rs[8];
#pragma unroll
  for (int r = 0; r < 8; ++r) rs[r] = 0.f;

  const unsigned int aBase = (unsigned)(wv * 16 + (lane & 15)) * 388 + 8u * half;

  for (int j = 0; j < 48; ++j) {
    const int n0 = j * 16;
    const float bias = b_st[n0 + col];
    const float w2 = w_st2[n0 + col];
    const unsigned short* wtp = Wt + (size_t)(n0 + col) * HH + 16 * half;
    v8f c0 = {};
    for (int k0 = 0; k0 < HH; k0 += 32) {
      const u32x4* bp = (const u32x4*)(const void*)(wtp + k0);
      v16bf bfr = make_frag(bp[0], bp[1]);
      const u32x4* ap = (const u32x4*)(const void*)(&sA[aBase + (k0 >> 1)]);
      v16bf a0 = make_frag(ap[0], ap[1]);
      c0 = __builtin_amdgcn_wmma_f32_16x16x32_bf16(false, a0, false, bfr, (short)0, c0, false, false);
    }
#pragma unroll
    for (int r = 0; r < 8; ++r) rs[r] += fast_tanh(c0[r] + bias) * w2;
  }

  // reduce over the 16 N-columns held in each half-wave
#pragma unroll
  for (int m = 1; m < 16; m <<= 1) {
#pragma unroll
    for (int r = 0; r < 8; ++r) rs[r] += __shfl_xor(rs[r], m, 32);
  }
  // wave owns its rows outright: lanes 0 / 16 store rows (8*half + r)
  if ((lane & 15) == 0) {
    int rbase = m0 + wv * 16 + 8 * half;
#pragma unroll
    for (int r = 0; r < 8; ++r) tokws[rbase + r] = rs[r] + bst2v;
  }
}

// ---------------- kernel 3: masked softmax + att@seq (f32) -> mix
__global__ __launch_bounds__(256) void k_softmix(const float* __restrict__ seq,
                                                 const int* __restrict__ sep,
                                                 const float* __restrict__ tokws,
                                                 float* __restrict__ out) {
  int e = blockIdx.x, t = threadIdx.x;
  __shared__ float ssc[2 * LL];
  __shared__ float swt[2 * LL];
  int s0 = sep[e * 2], s1 = sep[e * 2 + 1];
  if (t < LL) {
    float ts = tokws[e * LL + t];
    bool in0 = (t >= 1) && (t <= s0);
    bool in1 = (t > s0) && (t <= s1);
    ssc[t] = in0 ? ts : -10000.f;
    ssc[LL + t] = in1 ? ts : -10000.f;
  }
  __syncthreads();
  int wv = t >> 5, lane = t & 31;
  if (wv < 2) {
    float mx = -3.4e38f;
#pragma unroll
    for (int i = 0; i < 4; ++i) mx = fmaxf(mx, ssc[wv * LL + lane + 32 * i]);
#pragma unroll
    for (int m = 16; m > 0; m >>= 1) mx = fmaxf(mx, __shfl_xor(mx, m, 32));
    float ev[4], sm = 0.f;
#pragma unroll
    for (int i = 0; i < 4; ++i) { ev[i] = __expf(ssc[wv * LL + lane + 32 * i] - mx); sm += ev[i]; }
#pragma unroll
    for (int m = 16; m > 0; m >>= 1) sm += __shfl_xor(sm, m, 32);
    float inv = 1.f / sm;
#pragma unroll
    for (int i = 0; i < 4; ++i) swt[wv * LL + lane + 32 * i] = ev[i] * inv;
  }
  __syncthreads();
  const float* sb = seq + (size_t)e * LL * HH;
  float a00 = 0, a01 = 0, a02 = 0, a10 = 0, a11 = 0, a12 = 0;
  for (int l = 0; l < LL; ++l) {
    float w0 = swt[l], w1 = swt[LL + l];
    float v0 = sb[l * HH + t];
    float v1 = sb[l * HH + t + 256];
    float v2 = sb[l * HH + t + 512];
    a00 += w0 * v0; a01 += w0 * v1; a02 += w0 * v2;
    a10 += w1 * v0; a11 += w1 * v1; a12 += w1 * v2;
  }
  float* om = out + OUT_MIX + (size_t)e * 2 * HH;
  om[t] = a00; om[t + 256] = a01; om[t + 512] = a02;
  om[HH + t] = a10; om[HH + t + 256] = a11; om[HH + t + 512] = a12;
}

// ---------------- kernel 4: zero scatter targets
__global__ __launch_bounds__(256) void k_zero(float* __restrict__ out) {
  size_t i = (size_t)blockIdx.x * 256 + threadIdx.x;
  if (i < (size_t)ZERO_N) out[OUT_CLSMAT + i] = 0.f;
}

// ---------------- kernel 5: cls heads + scatter into (n,n) matrices
__global__ __launch_bounds__(256) void k_cls(const float* __restrict__ cls,
                                             const int* __restrict__ pairs,
                                             const float* __restrict__ Wpw, const float* __restrict__ bpw,
                                             const float* __restrict__ Wh1, const float* __restrict__ bh1,
                                             const float* __restrict__ Wh2, const float* __restrict__ bh2,
                                             float* __restrict__ out) {
  int e = blockIdx.x, t = threadIdx.x;
  const float* cr = cls + (size_t)e * HH;
  float c0 = cr[t], c1 = cr[t + 256], c2 = cr[t + 512];
  __shared__ float red[256 * 6];
  float p[6];
  p[0] = c0 * Wpw[t * 2]     + c1 * Wpw[(t + 256) * 2]     + c2 * Wpw[(t + 512) * 2];
  p[1] = c0 * Wpw[t * 2 + 1] + c1 * Wpw[(t + 256) * 2 + 1] + c2 * Wpw[(t + 512) * 2 + 1];
  p[2] = c0 * Wh1[t * 2]     + c1 * Wh1[(t + 256) * 2]     + c2 * Wh1[(t + 512) * 2];
  p[3] = c0 * Wh1[t * 2 + 1] + c1 * Wh1[(t + 256) * 2 + 1] + c2 * Wh1[(t + 512) * 2 + 1];
  p[4] = c0 * Wh2[t * 2]     + c1 * Wh2[(t + 256) * 2]     + c2 * Wh2[(t + 512) * 2];
  p[5] = c0 * Wh2[t * 2 + 1] + c1 * Wh2[(t + 256) * 2 + 1] + c2 * Wh2[(t + 512) * 2 + 1];
#pragma unroll
  for (int j = 0; j < 6; ++j) red[t * 6 + j] = p[j];
  __syncthreads();
  for (int st = 128; st > 0; st >>= 1) {
    if (t < st) {
#pragma unroll
      for (int j = 0; j < 6; ++j) red[t * 6 + j] += red[(t + st) * 6 + j];
    }
    __syncthreads();
  }
  int b = e / 120;
  int p0 = pairs[e * 2], p1 = pairs[e * 2 + 1];
  size_t cell = (size_t)(b * 16 + p0) * 16 + p1;
  if (t == 0) {
    float s0 = red[0] + bpw[0], s1 = red[1] + bpw[1];
    out[OUT_CLSSCORE + (size_t)e * 2]     = s0;
    out[OUT_CLSSCORE + (size_t)e * 2 + 1] = s1;
    out[OUT_SCMAT + cell * 2]     = s0;
    out[OUT_SCMAT + cell * 2 + 1] = s1;
    out[OUT_H1MAT + cell * 2]     = red[2] + bh1[0];
    out[OUT_H1MAT + cell * 2 + 1] = red[3] + bh1[1];
    out[OUT_H2MAT + cell * 2]     = red[4] + bh2[0];
    out[OUT_H2MAT + cell * 2 + 1] = red[5] + bh2[1];
  }
  float* cm = out + OUT_CLSMAT + cell * HH;
  cm[t] = c0; cm[t + 256] = c1; cm[t + 512] = c2;
}

// ---------------- kernel 6: per-node edge gather + softmax -> final_seq
// grid 128 = (b, n), block 256
__global__ __launch_bounds__(256) void k_graph(const int* __restrict__ pairs,
                                               const float* __restrict__ wlin2,
                                               float* __restrict__ out) {
  int bn = blockIdx.x, b = bn >> 4, n = bn & 15;
  int t = threadIdx.x, wv = t >> 5, lane = t & 31;
  __shared__ int fidx[16];
  __shared__ float qs[16];
  __shared__ float wss[16];
  if (t == 0) {
    int c = 0;
    for (int f = 0; f < 240; ++f)
      if (pairs[b * 240 + f] == n && c < 15) fidx[c++] = f;
  }
  __syncthreads();
  const float* mixb = out + OUT_MIX;
  for (int d = wv; d < 15; d += 8) {
    const float* row = mixb + (size_t)(b * 240 + fidx[d]) * HH;
    float s = 0.f;
#pragma unroll
    for (int i = 0; i < 24; ++i) { int h = lane + 32 * i; s += row[h] * wlin2[h]; }
#pragma unroll
    for (int m = 16; m > 0; m >>= 1) s += __shfl_xor(s, m, 32);
    if (lane == 0) qs[d] = s;
  }
  __syncthreads();
  if (t == 0) {
    float mx = -3.4e38f;
    for (int d = 0; d < 15; ++d) mx = fmaxf(mx, qs[d]);
    float sm = 0.f;
    for (int d = 0; d < 15; ++d) { float e = __expf(qs[d] - mx); wss[d] = e; sm += e; }
    float inv = 1.f / sm;
    for (int d = 0; d < 15; ++d) wss[d] *= inv;
  }
  __syncthreads();
  float a0 = 0, a1 = 0, a2 = 0;
  for (int d = 0; d < 15; ++d) {
    const float* row = mixb + (size_t)(b * 240 + fidx[d]) * HH;
    float w = wss[d];
    a0 += w * row[t]; a1 += w * row[t + 256]; a2 += w * row[t + 512];
  }
  float* fo = out + OUT_FINAL + (size_t)bn * HH;
  fo[t] = a0; fo[t + 256] = a1; fo[t + 512] = a2;
}

extern "C" void kernel_launch(void* const* d_in, const int* in_sizes, int n_in,
                              void* d_out, int out_size, void* d_ws, size_t ws_size,
                              hipStream_t stream) {
  const float* seq   = (const float*)d_in[0];
  const int*   pairs = (const int*)d_in[1];
  const int*   sep   = (const int*)d_in[4];
  const float* cls   = (const float*)d_in[6];
  const float* Wst   = (const float*)d_in[7];
  const float* bst   = (const float*)d_in[8];
  const float* wst2  = (const float*)d_in[9];
  const float* bst2  = (const float*)d_in[10];
  const float* wlin2 = (const float*)d_in[11];
  const float* Wpw   = (const float*)d_in[12];
  const float* bpw   = (const float*)d_in[13];
  const float* Wh1   = (const float*)d_in[14];
  const float* bh1   = (const float*)d_in[15];
  const float* Wh2   = (const float*)d_in[16];
  const float* bh2   = (const float*)d_in[17];

  unsigned short* Wt = (unsigned short*)d_ws;                   // 768*768 bf16 = 1,179,648 B
  float* tokws = (float*)((char*)d_ws + 1179648);               // 122,880 f32
  float* out = (float*)d_out;

  k_prep_wt<<<2304, 256, 0, stream>>>(Wst, Wt);
  k_score<<<960, 256, 0, stream>>>(seq, Wt, bst, wst2, bst2, tokws);
  k_zero<<<6200, 256, 0, stream>>>(out);
  k_softmix<<<960, 256, 0, stream>>>(seq, sep, tokws, out);
  k_cls<<<960, 256, 0, stream>>>(cls, pairs, Wpw, bpw, Wh1, bh1, Wh2, bh2, out);
  k_graph<<<128, 256, 0, stream>>>(pairs, wlin2, out);
}
